// HeteroGPSConv_11527692223016
// MI455X (gfx1250) — matrix operand order repaired and checked by reading, not compiled, not run on hardware
//
// HeteroGPSConv forward for MI455X (gfx1250, wave32, WMMA).
//
// Design notes (compile-only; reasoned from MI455X specs):
//  * All dense 128-wide linears run through v_wmma_f32_16x16x32_bf16 with f32
//    accumulation. f32 WMMA is 16x16x4 (8x less K/instr); bf16 keeps f32's
//    exponent range and weights are ~N(0, 0.05) -> bf16 A/B, f32 C is the
//    throughput-correct precision on CDNA5.
//  * The model is memory/atomic bound (~150 GFLOP vs few-GB traffic @23.3TB/s),
//    so bias/LayerNorm/ReLU/residual/sigmoid-gate epilogues are fused into the
//    GEMMs (one 16-row slab per workgroup owns the full 128-dim row -> LN done
//    with LDS + 16-lane shfl_xor reductions, coalesced float4 stores).
//  * Weights pre-packed once to WMMA B-fragment order (bf16) -> each wave loads
//    its B fragments as contiguous 32B/lane global loads.
//  * Graph irregularity via native f32 atomics: GINE segment-sum with
//    global_atomic_add_f32; HGT segment-softmax two-pass (atomic-max logits,
//    then exp/atomic-add of den and value-weighted num).
//  * HGT relation einsums (1.2M edges x 4 heads x 32x32) are gather-WMMAs:
//    16 edges/wave, K rows gathered by source index into A fragments.
//
// Param flattening assumption: inputs 0..7 are the data tensors in
// setup_inputs() insertion order; d_in[8..112] are params leaves in JAX pytree
// order (nested dicts flattened by sorted key). Index table in code below.

#include <hip/hip_runtime.h>
#include <math.h>

typedef __attribute__((ext_vector_type(16))) __bf16          v16bf;
typedef __attribute__((ext_vector_type(8)))  float           v8f;
typedef __attribute__((ext_vector_type(16))) unsigned short  v16u;

#define DEVINL __device__ __forceinline__

DEVINL unsigned short f2bf(float f) {          // f32 -> bf16 RNE
  unsigned u = __float_as_uint(f);
  u += 0x7FFFu + ((u >> 16) & 1u);
  return (unsigned short)(u >> 16);
}

DEVINL float geluf(float v) {                  // exact gelu (approximate=False)
  return 0.5f * v * (1.0f + erff(v * 0.70710678118654752440f));
}

DEVINL v8f wmma_bf16(v16bf a, v16bf b, v8f c) {
  return __builtin_amdgcn_wmma_f32_16x16x32_bf16(false, a, false, b, (short)0, c,
                                                 false, false);
}

DEVINL void atomicMaxF(float* addr, float v) { // float atomic max via int trick
  if (v >= 0.0f) atomicMax((int*)addr, __float_as_int(v));
  else           atomicMin((unsigned int*)addr, __float_as_uint(v));
}

// ---------------------------------------------------------------------------
// Weight packing: f32 [K=128][N=128] row-major -> bf16 WMMA B-fragment order.
// Fragment f = tn*4 + tk (tn: 16-col tile, tk: 32-K tile); within a fragment
// lane l element i holds B[k][n], k = tk*32 + (l>=16?8:0) + (i<8?i:i+8), n = tn*16+(l&15)
// ---------------------------------------------------------------------------
struct Ptr28 { const float* p[28]; };
struct Ptr6  { const float* p[6]; };

__global__ __launch_bounds__(256) void pack_dense_kernel(Ptr28 srcs, __bf16* dst) {
  const float* src = srcs.p[blockIdx.x];
  __bf16* d = dst + (long)blockIdx.x * 16384;
  for (int o = threadIdx.x; o < 16384; o += 256) {
    int f = o >> 9, l = (o >> 4) & 31, i = o & 15;
    int tn = f >> 2, tk = f & 3;
    int k = tk * 32 + ((l >= 16) ? 8 : 0) + (i < 8 ? i : i + 8);
    int n = tn * 16 + (l & 15);
    d[o] = __builtin_bit_cast(__bf16, f2bf(src[k * 128 + n]));
  }
}

// rel matrices: [4 heads][32][32]; per head 2 column tiles, single K tile.
// fragment f = h*2 + tn
__global__ __launch_bounds__(256) void pack_rel_kernel(Ptr6 srcs, __bf16* dst) {
  const float* src = srcs.p[blockIdx.x];
  __bf16* d = dst + (long)blockIdx.x * 4096;
  for (int o = threadIdx.x; o < 4096; o += 256) {
    int f = o >> 9, l = (o >> 4) & 31, i = o & 15;
    int h = f >> 1, tn = f & 1;
    int k = ((l >= 16) ? 8 : 0) + (i < 8 ? i : i + 8);
    int n = tn * 16 + (l & 15);
    d[o] = __builtin_bit_cast(__bf16, f2bf(src[(h * 32 + k) * 32 + n]));
  }
}

// ---------------------------------------------------------------------------
// fill
// ---------------------------------------------------------------------------
__global__ void fill_kernel(float* p, long n, float v) {
  long i = (long)blockIdx.x * blockDim.x + threadIdx.x;
  long stride = (long)gridDim.x * blockDim.x;
  for (; i < n; i += stride) p[i] = v;
}

// ---------------------------------------------------------------------------
// GINE edge aggregation: aggr[di] += relu(x[si] + ea[e]*We + be)
// one thread = (edge, 4 channels); f32 VMEM atomics
// ---------------------------------------------------------------------------
__global__ __launch_bounds__(256) void gine_scatter_kernel(
    const float* __restrict__ X, const float* __restrict__ ea,
    const int* __restrict__ si, const int* __restrict__ di,
    const float* __restrict__ We, const float* __restrict__ be,
    float* __restrict__ aggr, long E) {
  long t = (long)blockIdx.x * 256 + threadIdx.x;
  long e = t >> 5;
  if (e >= E) return;
  int j0 = (int)(t & 31) * 4;
  float a = ea[e];
  long s = si[e], d = di[e];
  const float* xp = X + s * 128 + j0;
  float* ap = aggr + d * 128 + j0;
#pragma unroll
  for (int k = 0; k < 4; k++) {
    float v = xp[k] + a * We[j0 + k] + be[j0 + k];
    v = fmaxf(v, 0.0f);
    unsafeAtomicAdd(&ap[k], v);
  }
}

// ---------------------------------------------------------------------------
// Fused GEMM: Y[16-row slab] = epilogue( A(X...) @ W + b )
// 256 thr = 8 waves; wave w owns output columns [16w,16w+16); K-loop of 4 WMMAs
// AMODE: 0: A=X1 | 1: A=X1+X2 | 2: A=gelu(X1/(X2den+1e-16)) | 3: A=X1+X2+X3
// MODE : 0: +b | 1: relu(LN(+b)) [+=Y if acc] | 2: relu(+b)
//        3: LN(relu(+b)+R)       | 4: al*(+b) + (1-al)*R, al=sigmoid(skip)
// ---------------------------------------------------------------------------
template <int AMODE, int MODE>
__global__ __launch_bounds__(256) void gemm128_kernel(
    const float* __restrict__ X1, const float* __restrict__ X2,
    const float* __restrict__ X3, const __bf16* __restrict__ Wp,
    const float* __restrict__ bias, const float* __restrict__ gamma,
    const float* __restrict__ betaLN, const float* __restrict__ R,
    const float* __restrict__ skipP, float acc, float* __restrict__ Y) {
  __shared__ unsigned short lA[16 * 136];  // bf16 A slab, padded vs bank conflicts
  __shared__ float lC[16 * 132];           // f32 C slab

  const int t = threadIdx.x;
  const long rb = (long)blockIdx.x * 16;
  const int row = t >> 4;
  const int j0 = (t & 15) * 8;
  const long gr = rb + row;

  { // stage A (16 x 128 f32 -> bf16 in LDS), fusing input-side ops
    const float* x1p = X1 + gr * 128 + j0;
    float v[8];
#pragma unroll
    for (int jj = 0; jj < 8; jj++) v[jj] = x1p[jj];
    if constexpr (AMODE == 1 || AMODE == 3) {
      const float* x2p = X2 + gr * 128 + j0;
#pragma unroll
      for (int jj = 0; jj < 8; jj++) v[jj] += x2p[jj];
    }
    if constexpr (AMODE == 3) {
      const float* x3p = X3 + gr * 128 + j0;
#pragma unroll
      for (int jj = 0; jj < 8; jj++) v[jj] += x3p[jj];
    }
    if constexpr (AMODE == 2) {
      float dinv = 1.0f / (X2[gr * 4 + (j0 >> 5)] + 1e-16f);
#pragma unroll
      for (int jj = 0; jj < 8; jj++) v[jj] = geluf(v[jj] * dinv);
    }
#pragma unroll
    for (int jj = 0; jj < 8; jj++) lA[row * 136 + j0 + jj] = f2bf(v[jj]);
  }
  __syncthreads();

  const int l = t & 31, w = t >> 5;
  const int m = l & 15, hi = l >> 4;
  v8f c = {};
#pragma unroll
  for (int tk = 0; tk < 4; tk++) {
    v16u ar;
#pragma unroll
    for (int i = 0; i < 16; i++) {  // A layout: CDNA5 ISA 16-bit 16x32
      int k = tk * 32 + hi * 8 + (i < 8 ? i : i + 8);
      ar[i] = lA[m * 136 + k];
    }
    v16bf a = __builtin_bit_cast(v16bf, ar);
    v16bf b = *(const v16bf*)(Wp + ((w * 4 + tk) * 512 + l * 16));
    c = wmma_bf16(a, b, c);
  }
#pragma unroll
  for (int r = 0; r < 8; r++)  // C layout: VGPR r -> row r+8*hi, col = lane&15
    lC[(r + hi * 8) * 132 + w * 16 + m] = c[r];
  __syncthreads();

  // epilogue: thread t handles (row, cols j0..j0+7); 16 threads per row
  float b8[8], r8[8], u8[8], o8[8];
#pragma unroll
  for (int jj = 0; jj < 8; jj++) b8[jj] = bias[j0 + jj];
  if constexpr (MODE == 3 || MODE == 4) {
#pragma unroll
    for (int jj = 0; jj < 8; jj++) r8[jj] = R[gr * 128 + j0 + jj];
  }
#pragma unroll
  for (int jj = 0; jj < 8; jj++) {
    float v = lC[row * 132 + j0 + jj] + b8[jj];
    if constexpr (MODE == 3) v = fmaxf(v, 0.0f) + r8[jj];
    u8[jj] = v;
  }
  if constexpr (MODE == 1 || MODE == 3) {
    float s1 = 0.f, s2 = 0.f;
#pragma unroll
    for (int jj = 0; jj < 8; jj++) { s1 += u8[jj]; s2 += u8[jj] * u8[jj]; }
#pragma unroll
    for (int d = 1; d < 16; d <<= 1) {  // reduce across the 16 lanes of a row
      s1 += __shfl_xor(s1, d, 32);
      s2 += __shfl_xor(s2, d, 32);
    }
    float mean = s1 * (1.0f / 128.0f);
    float var = s2 * (1.0f / 128.0f) - mean * mean;
    float rstd = rsqrtf(var + 1e-5f);
#pragma unroll
    for (int jj = 0; jj < 8; jj++) {
      float o = (u8[jj] - mean) * rstd * gamma[j0 + jj] + betaLN[j0 + jj];
      if constexpr (MODE == 1) {
        o = fmaxf(o, 0.0f);
        if (acc != 0.0f) o += Y[gr * 128 + j0 + jj];
      }
      o8[jj] = o;
    }
  } else if constexpr (MODE == 0) {
#pragma unroll
    for (int jj = 0; jj < 8; jj++) o8[jj] = u8[jj];
  } else if constexpr (MODE == 2) {
#pragma unroll
    for (int jj = 0; jj < 8; jj++) o8[jj] = fmaxf(u8[jj], 0.0f);
  } else if constexpr (MODE == 4) {
    float al = 1.0f / (1.0f + expf(-skipP[0]));
#pragma unroll
    for (int jj = 0; jj < 8; jj++) o8[jj] = al * u8[jj] + (1.0f - al) * r8[jj];
  }
  float4* yp = reinterpret_cast<float4*>(Y + gr * 128 + j0);
  yp[0] = make_float4(o8[0], o8[1], o8[2], o8[3]);
  yp[1] = make_float4(o8[4], o8[5], o8[6], o8[7]);
}

// ---------------------------------------------------------------------------
// HGT attention pass 1: ke = gather(K,si) @ arel per head (WMMA), then
// logits a[e,h] = <q[di], ke> * prel/sqrt(32); store + atomic segment max.
// 128 thr = 4 waves x 16 edges.
// ---------------------------------------------------------------------------
__global__ __launch_bounds__(128) void hgt_att1_kernel(
    const float* __restrict__ Kmat, const float* __restrict__ Qmat,
    const int* __restrict__ si, const int* __restrict__ di,
    const __bf16* __restrict__ arelp, const float* __restrict__ prel,
    float* __restrict__ Lout, float* __restrict__ mx, int E) {
  __shared__ float lKE[4 * 4 * 16 * 32];  // [wave][head][m][f]
  const int t = threadIdx.x, l = t & 31, w = t >> 5;
  const int m = l & 15, hi = l >> 4;
  long e = (long)blockIdx.x * 64 + w * 16 + m;
  if (e >= E) e = E - 1;  // clamped duplicate; results discarded in phase 2
  const float* kp = Kmat + (long)si[e] * 128;
#pragma unroll
  for (int h = 0; h < 4; h++) {
    v16u ar;
#pragma unroll
    for (int i = 0; i < 16; i++) {
      int k = hi * 8 + (i < 8 ? i : i + 8);
      ar[i] = f2bf(kp[h * 32 + k]);
    }
    v16bf a = __builtin_bit_cast(v16bf, ar);
    v16bf b0 = *(const v16bf*)(arelp + (h * 2 + 0) * 512 + l * 16);
    v16bf b1 = *(const v16bf*)(arelp + (h * 2 + 1) * 512 + l * 16);
    v8f c0 = {}, c1 = {};
    c0 = wmma_bf16(a, b0, c0);
    c1 = wmma_bf16(a, b1, c1);
#pragma unroll
    for (int r = 0; r < 8; r++) {
      int mm = r + hi * 8;
      lKE[((w * 4 + h) * 16 + mm) * 32 + m]      = c0[r];
      lKE[((w * 4 + h) * 16 + mm) * 32 + 16 + m] = c1[r];
    }
  }
  __syncthreads();
  int idx = t >> 1, h0 = (t & 1) * 2;
  long e2 = (long)blockIdx.x * 64 + idx;
  if (e2 < E) {
    long drow = di[e2];
    const float* qp = Qmat + drow * 128;
    int wv = idx >> 4, mm = idx & 15;
#pragma unroll
    for (int hh = 0; hh < 2; hh++) {
      int h = h0 + hh;
      const float* ke = &lKE[((wv * 4 + h) * 16 + mm) * 32];
      float s = 0.f;
#pragma unroll
      for (int f = 0; f < 32; f++) s += qp[h * 32 + f] * ke[f];
      float a = s * prel[h] * 0.17677669529663687f;  // 1/sqrt(32)
      Lout[e2 * 4 + h] = a;
      atomicMaxF(&mx[drow * 4 + h], a);
    }
  }
}

// ---------------------------------------------------------------------------
// HGT attention pass 2: ve = gather(V,si) @ mrel (WMMA), wgt = exp(L - mx),
// atomic den += wgt, num += wgt * ve.
// ---------------------------------------------------------------------------
__global__ __launch_bounds__(128) void hgt_att2_kernel(
    const float* __restrict__ Vmat, const int* __restrict__ si,
    const int* __restrict__ di, const __bf16* __restrict__ mrelp,
    const float* __restrict__ Lin, const float* __restrict__ mx,
    float* __restrict__ den, float* __restrict__ num, int E) {
  __shared__ float lVE[4 * 4 * 16 * 32];
  const int t = threadIdx.x, l = t & 31, w = t >> 5;
  const int m = l & 15, hi = l >> 4;
  long e = (long)blockIdx.x * 64 + w * 16 + m;
  if (e >= E) e = E - 1;
  const float* vp = Vmat + (long)si[e] * 128;
#pragma unroll
  for (int h = 0; h < 4; h++) {
    v16u ar;
#pragma unroll
    for (int i = 0; i < 16; i++) {
      int k = hi * 8 + (i < 8 ? i : i + 8);
      ar[i] = f2bf(vp[h * 32 + k]);
    }
    v16bf a = __builtin_bit_cast(v16bf, ar);
    v16bf b0 = *(const v16bf*)(mrelp + (h * 2 + 0) * 512 + l * 16);
    v16bf b1 = *(const v16bf*)(mrelp + (h * 2 + 1) * 512 + l * 16);
    v8f c0 = {}, c1 = {};
    c0 = wmma_bf16(a, b0, c0);
    c1 = wmma_bf16(a, b1, c1);
#pragma unroll
    for (int r = 0; r < 8; r++) {
      int mm = r + hi * 8;
      lVE[((w * 4 + h) * 16 + mm) * 32 + m]      = c0[r];
      lVE[((w * 4 + h) * 16 + mm) * 32 + 16 + m] = c1[r];
    }
  }
  __syncthreads();
  int idx = t >> 1, h0 = (t & 1) * 2;
  long e2 = (long)blockIdx.x * 64 + idx;
  if (e2 < E) {
    long drow = di[e2];
    int wv = idx >> 4, mm = idx & 15;
#pragma unroll
    for (int hh = 0; hh < 2; hh++) {
      int h = h0 + hh;
      float wgt = expf(Lin[e2 * 4 + h] - mx[drow * 4 + h]);
      unsafeAtomicAdd(&den[drow * 4 + h], wgt);
      float* np = num + drow * 128 + h * 32;
      const float* ve = &lVE[((wv * 4 + h) * 16 + mm) * 32];
#pragma unroll
      for (int f = 0; f < 32; f++) unsafeAtomicAdd(&np[f], wgt * ve[f]);
    }
  }
}

// ---------------------------------------------------------------------------
// Host orchestration
// ---------------------------------------------------------------------------
extern "C" void kernel_launch(void* const* d_in, const int* in_sizes, int n_in,
                              void* d_out, int out_size, void* d_ws,
                              size_t ws_size, hipStream_t stream) {
  (void)in_sizes; (void)n_in; (void)out_size; (void)ws_size;
  const long NELT = 100000L * 128;
  const int EC = 600000, EW = 300000, ER = 300000;
  const int GG = 6250;  // 100000/16 row slabs

  auto F = [&](int i) -> const float* { return (const float*)d_in[i]; };
  const float* x_paper  = F(0);
  const float* x_author = F(1);
  const int* ei_c = (const int*)d_in[2];  const float* ea_c = F(3);
  const int* ei_w = (const int*)d_in[4];  const float* ea_w = F(5);
  const int* ei_r = (const int*)d_in[6];  const float* ea_r = F(7);
  // Param flat indices (sorted pytree):
  //  gine.cites:  W1=8 W2=9 We=10 b1=11 b2=12 be=13 g1=14 g2=15 t1=16 t2=17
  //  gine.rev:    18..27   gine.writes: 28..37
  //  gine_post.author: W1=38 W2=39 b1=40 b2=41 g=42 t=43 ; .paper: 44..49
  //  gps_out.author: W=50 b=51 ; .paper: W=52 b=53
  //  gps_post.author: 54..59 ; .paper: 60..65
  //  hgt_node.author: Wa=66 Wk=67 Wq=68 Wv=69 ba=70 bk=71 bq=72 bv=73 skip=74
  //  hgt_node.paper : 75..83
  //  hgt_post.author: 84..89 ; .paper: 90..95
  //  hgt_rel.cites: arel=96 mrel=97 prel=98 ; rev: 99..101 ; writes: 102..104
  //  norm1.author: g=105 t=106 ; norm1.paper: 107,108
  //  norm2.author: 109,110 ; norm2.paper: 111,112

  // workspace carve (~590 MB)
  char* wsp = (char*)d_ws;
  auto carve = [&](size_t bytes) -> void* {
    void* p = (void*)wsp;
    wsp += (bytes + 255) & ~(size_t)255;
    return p;
  };
  __bf16* wpack   = (__bf16*)carve(28 * 16384 * sizeof(__bf16));
  __bf16* relpack = (__bf16*)carve(6 * 4096 * sizeof(__bf16));
  float* logits = (float*)carve((size_t)1200000 * 4 * sizeof(float));
  float* mxP  = (float*)carve(400000 * sizeof(float));
  float* mxA  = (float*)carve(400000 * sizeof(float));
  float* denP = (float*)carve(400000 * sizeof(float));
  float* denA = (float*)carve(400000 * sizeof(float));
  float* B[11];
  for (int i = 0; i < 11; i++) B[i] = (float*)carve((size_t)NELT * sizeof(float));
  float* out = (float*)d_out;

  auto WPp = [&](int s) -> const __bf16* { return wpack + (long)s * 16384; };
  auto RPp = [&](int s) -> const __bf16* { return relpack + (long)s * 4096; };

  // ---- pack weights into WMMA fragment order (bf16) ----
  Ptr28 p28;
  const int dlist[28] = {8, 9, 18, 19, 28, 29, 38, 39, 44, 45, 50, 52, 54, 55,
                         60, 61, 66, 67, 68, 69, 75, 76, 77, 78, 84, 85, 90, 91};
  for (int i = 0; i < 28; i++) p28.p[i] = F(dlist[i]);
  pack_dense_kernel<<<28, 256, 0, stream>>>(p28, wpack);
  Ptr6 p6;
  const int rlist[6] = {96, 97, 99, 100, 102, 103};
  for (int i = 0; i < 6; i++) p6.p[i] = F(rlist[i]);
  pack_rel_kernel<<<6, 256, 0, stream>>>(p6, relpack);

  // ---- GINE branch ----
  fill_kernel<<<1024, 256, 0, stream>>>(B[0], NELT, 0.f);  // aggr_cites
  fill_kernel<<<1024, 256, 0, stream>>>(B[1], NELT, 0.f);  // aggr_writes
  fill_kernel<<<1024, 256, 0, stream>>>(B[2], NELT, 0.f);  // aggr_rev
  gine_scatter_kernel<<<EC / 8, 256, 0, stream>>>(x_paper, ea_c, ei_c, ei_c + EC,
                                                  F(10), F(13), B[0], EC);
  gine_scatter_kernel<<<EW / 8, 256, 0, stream>>>(x_author, ea_w, ei_w, ei_w + EW,
                                                  F(30), F(33), B[1], EW);
  gine_scatter_kernel<<<ER / 8, 256, 0, stream>>>(x_paper, ea_r, ei_r, ei_r + ER,
                                                  F(20), F(23), B[2], ER);
  // gine_nn per edge type (summed into per-dst-type buffers B4/B5)
  gemm128_kernel<1, 1><<<GG, 256, 0, stream>>>(x_paper, B[0], nullptr, WPp(0),
      F(11), F(14), F(16), nullptr, nullptr, 0.f, B[3]);
  gemm128_kernel<0, 1><<<GG, 256, 0, stream>>>(B[3], nullptr, nullptr, WPp(1),
      F(12), F(15), F(17), nullptr, nullptr, 0.f, B[4]);
  gemm128_kernel<1, 1><<<GG, 256, 0, stream>>>(x_paper, B[1], nullptr, WPp(4),
      F(31), F(34), F(36), nullptr, nullptr, 0.f, B[3]);
  gemm128_kernel<0, 1><<<GG, 256, 0, stream>>>(B[3], nullptr, nullptr, WPp(5),
      F(32), F(35), F(37), nullptr, nullptr, 1.f, B[4]);  // accumulate
  gemm128_kernel<1, 1><<<GG, 256, 0, stream>>>(x_author, B[2], nullptr, WPp(2),
      F(21), F(24), F(26), nullptr, nullptr, 0.f, B[3]);
  gemm128_kernel<0, 1><<<GG, 256, 0, stream>>>(B[3], nullptr, nullptr, WPp(3),
      F(22), F(25), F(27), nullptr, nullptr, 0.f, B[5]);
  // gine_post + residual LN -> h1 (B6 paper, B7 author)
  gemm128_kernel<0, 1><<<GG, 256, 0, stream>>>(B[4], nullptr, nullptr, WPp(8),
      F(46), F(48), F(49), nullptr, nullptr, 0.f, B[3]);
  gemm128_kernel<0, 3><<<GG, 256, 0, stream>>>(B[3], nullptr, nullptr, WPp(9),
      F(47), F(107), F(108), x_paper, nullptr, 0.f, B[6]);
  gemm128_kernel<0, 1><<<GG, 256, 0, stream>>>(B[5], nullptr, nullptr, WPp(6),
      F(40), F(42), F(43), nullptr, nullptr, 0.f, B[3]);
  gemm128_kernel<0, 3><<<GG, 256, 0, stream>>>(B[3], nullptr, nullptr, WPp(7),
      F(41), F(105), F(106), x_author, nullptr, 0.f, B[7]);

  // ---- HGT branch: K/Q/V ----
  gemm128_kernel<0, 0><<<GG, 256, 0, stream>>>(x_paper, nullptr, nullptr, WPp(21),
      F(80), nullptr, nullptr, nullptr, nullptr, 0.f, B[0]);  // kP
  gemm128_kernel<0, 0><<<GG, 256, 0, stream>>>(x_paper, nullptr, nullptr, WPp(22),
      F(81), nullptr, nullptr, nullptr, nullptr, 0.f, B[1]);  // qP
  gemm128_kernel<0, 0><<<GG, 256, 0, stream>>>(x_paper, nullptr, nullptr, WPp(23),
      F(82), nullptr, nullptr, nullptr, nullptr, 0.f, B[2]);  // vP
  gemm128_kernel<0, 0><<<GG, 256, 0, stream>>>(x_author, nullptr, nullptr, WPp(17),
      F(71), nullptr, nullptr, nullptr, nullptr, 0.f, B[4]);  // kA
  gemm128_kernel<0, 0><<<GG, 256, 0, stream>>>(x_author, nullptr, nullptr, WPp(18),
      F(72), nullptr, nullptr, nullptr, nullptr, 0.f, B[5]);  // qA
  gemm128_kernel<0, 0><<<GG, 256, 0, stream>>>(x_author, nullptr, nullptr, WPp(19),
      F(73), nullptr, nullptr, nullptr, nullptr, 0.f, B[8]);  // vA

  // segment-softmax state
  fill_kernel<<<256, 256, 0, stream>>>(mxP, 400000, -INFINITY);
  fill_kernel<<<256, 256, 0, stream>>>(mxA, 400000, -INFINITY);
  fill_kernel<<<256, 256, 0, stream>>>(denP, 400000, 0.f);
  fill_kernel<<<256, 256, 0, stream>>>(denA, 400000, 0.f);
  fill_kernel<<<1024, 256, 0, stream>>>(B[9], NELT, 0.f);   // numP
  fill_kernel<<<1024, 256, 0, stream>>>(B[10], NELT, 0.f);  // numA

  float* Lc = logits;
  float* Lw = logits + (long)EC * 4;
  float* Lr = logits + (long)(EC + EW) * 4;
  hgt_att1_kernel<<<(EC + 63) / 64, 128, 0, stream>>>(B[0], B[1], ei_c, ei_c + EC,
      RPp(0), F(98), Lc, mxP, EC);
  hgt_att1_kernel<<<(EW + 63) / 64, 128, 0, stream>>>(B[4], B[1], ei_w, ei_w + EW,
      RPp(4), F(104), Lw, mxP, EW);
  hgt_att1_kernel<<<(ER + 63) / 64, 128, 0, stream>>>(B[0], B[5], ei_r, ei_r + ER,
      RPp(2), F(101), Lr, mxA, ER);
  hgt_att2_kernel<<<(EC + 63) / 64, 128, 0, stream>>>(B[2], ei_c, ei_c + EC,
      RPp(1), Lc, mxP, denP, B[9], EC);
  hgt_att2_kernel<<<(EW + 63) / 64, 128, 0, stream>>>(B[8], ei_w, ei_w + EW,
      RPp(5), Lw, mxP, denP, B[9], EW);
  hgt_att2_kernel<<<(ER + 63) / 64, 128, 0, stream>>>(B[2], ei_r, ei_r + ER,
      RPp(3), Lr, mxA, denA, B[10], ER);

  // o = gelu(num/den) @ Wa + ba ; hgt = sigmoid(skip)*o + (1-sigmoid)*x
  gemm128_kernel<2, 4><<<GG, 256, 0, stream>>>(B[9], denP, nullptr, WPp(20),
      F(79), nullptr, nullptr, x_paper, F(83), 0.f, B[0]);   // hgtP
  gemm128_kernel<2, 4><<<GG, 256, 0, stream>>>(B[10], denA, nullptr, WPp(16),
      F(70), nullptr, nullptr, x_author, F(74), 0.f, B[4]);  // hgtA
  // hgt_post + residual LN -> h2 (B9 paper, B10 author)
  gemm128_kernel<0, 1><<<GG, 256, 0, stream>>>(B[0], nullptr, nullptr, WPp(26),
      F(92), F(94), F(95), nullptr, nullptr, 0.f, B[3]);
  gemm128_kernel<0, 3><<<GG, 256, 0, stream>>>(B[3], nullptr, nullptr, WPp(27),
      F(93), F(111), F(112), x_paper, nullptr, 0.f, B[9]);
  gemm128_kernel<0, 1><<<GG, 256, 0, stream>>>(B[4], nullptr, nullptr, WPp(24),
      F(86), F(88), F(89), nullptr, nullptr, 0.f, B[3]);
  gemm128_kernel<0, 3><<<GG, 256, 0, stream>>>(B[3], nullptr, nullptr, WPp(25),
      F(87), F(109), F(110), x_author, nullptr, 0.f, B[10]);

  // ---- combine: s = h1+h2 ; gps_post ; out = relu((post+s)@W + b) ----
  gemm128_kernel<1, 1><<<GG, 256, 0, stream>>>(B[6], B[9], nullptr, WPp(14),
      F(62), F(64), F(65), nullptr, nullptr, 0.f, B[3]);
  gemm128_kernel<0, 0><<<GG, 256, 0, stream>>>(B[3], nullptr, nullptr, WPp(15),
      F(63), nullptr, nullptr, nullptr, nullptr, 0.f, B[2]);
  gemm128_kernel<3, 2><<<GG, 256, 0, stream>>>(B[2], B[6], B[9], WPp(11),
      F(53), nullptr, nullptr, nullptr, nullptr, 0.f, out);          // paper
  gemm128_kernel<1, 1><<<GG, 256, 0, stream>>>(B[7], B[10], nullptr, WPp(12),
      F(56), F(58), F(59), nullptr, nullptr, 0.f, B[3]);
  gemm128_kernel<0, 0><<<GG, 256, 0, stream>>>(B[3], nullptr, nullptr, WPp(13),
      F(57), nullptr, nullptr, nullptr, nullptr, 0.f, B[2]);
  gemm128_kernel<3, 2><<<GG, 256, 0, stream>>>(B[2], B[7], B[10], WPp(10),
      F(51), nullptr, nullptr, nullptr, nullptr, 0.f, out + NELT);   // author
}
